// AttentionBlock_85581518340782
// MI455X (gfx1250) — matrix operand order repaired and checked by reading, not compiled, not run on hardware
//
#include <hip/hip_runtime.h>
#include <hip/hip_bf16.h>

typedef __attribute__((ext_vector_type(16))) __bf16 bf16x16;
typedef __attribute__((ext_vector_type(8)))  float  v8f;
typedef __attribute__((ext_vector_type(4)))  int    i32x4;

union Frag16 { bf16x16 v; uint4 u[2]; };

__device__ __forceinline__ unsigned short f32_to_bf16(float f) {
    unsigned int u = __float_as_uint(f);
    u += 0x7FFFu + ((u >> 16) & 1u);   // round to nearest even
    return (unsigned short)(u >> 16);
}

__device__ __forceinline__ v8f wmma_bf16(const bf16x16& a, const bf16x16& b, const v8f& c) {
    return __builtin_amdgcn_wmma_f32_16x16x32_bf16(false, a, false, b, (short)0, c, false, false);
}

// ---------------------------------------------------------------------------
// CDNA5 async global->LDS copy (ASYNCcnt path), builtin if present else asm.
// Builtin signature (probe-confirmed by diagnostic): (v4i AS1*, v4i AS3*, Ii, Ii)
// ---------------------------------------------------------------------------
typedef __attribute__((address_space(1))) i32x4* as1_i32x4_p;
typedef __attribute__((address_space(3))) i32x4* as3_i32x4_p;

__device__ __forceinline__ void async_copy_b128(const unsigned short* g, unsigned short* l) {
#if __has_builtin(__builtin_amdgcn_global_load_async_to_lds_b128)
    __builtin_amdgcn_global_load_async_to_lds_b128(
        (as1_i32x4_p)(unsigned short*)g, (as3_i32x4_p)l, 0, 0);
#else
    unsigned loff = (unsigned)(size_t)(__attribute__((address_space(3))) unsigned short*)l;
    asm volatile("global_load_async_to_lds_b128 %0, %1, off"
                 :: "v"(loff), "v"(g) : "memory");
#endif
}

template <int N>
__device__ __forceinline__ void wait_asynccnt() {
#if __has_builtin(__builtin_amdgcn_s_wait_asynccnt)
    __builtin_amdgcn_s_wait_asynccnt((unsigned short)N);
#else
    asm volatile("s_wait_asynccnt %0" :: "i"(N) : "memory");
#endif
}

#define D_HID 1024
#define NBATCH 4
#define SEQ 2048
#define NHEAD 16
#define DHEAD 64

// ---------------------------------------------------------------------------
// LayerNorm: one block per row, 256 threads, 4 elements/thread, bf16 output.
// ---------------------------------------------------------------------------
__global__ __launch_bounds__(256) void ln_kernel(
    const float* __restrict__ x, const float* __restrict__ gamma,
    const float* __restrict__ beta, unsigned short* __restrict__ xn)
{
    __shared__ float red[256];
    const int t = threadIdx.x;
    const size_t base = (size_t)blockIdx.x * D_HID;
    float4 v = ((const float4*)(x + base))[t];

    red[t] = v.x + v.y + v.z + v.w;
    __syncthreads();
    for (int s = 128; s > 0; s >>= 1) {
        if (t < s) red[t] += red[t + s];
        __syncthreads();
    }
    const float mean = red[0] * (1.0f / D_HID);
    __syncthreads();

    float dx = v.x - mean, dy = v.y - mean, dz = v.z - mean, dw = v.w - mean;
    red[t] = dx*dx + dy*dy + dz*dz + dw*dw;
    __syncthreads();
    for (int s = 128; s > 0; s >>= 1) {
        if (t < s) red[t] += red[t + s];
        __syncthreads();
    }
    const float rstd = rsqrtf(red[0] * (1.0f / D_HID) + 1e-5f);

    const int c = t * 4;
    float o0 = dx * rstd * gamma[c+0] + beta[c+0];
    float o1 = dy * rstd * gamma[c+1] + beta[c+1];
    float o2 = dz * rstd * gamma[c+2] + beta[c+2];
    float o3 = dw * rstd * gamma[c+3] + beta[c+3];
    xn[base + c + 0] = f32_to_bf16(o0);
    xn[base + c + 1] = f32_to_bf16(o1);
    xn[base + c + 2] = f32_to_bf16(o2);
    xn[base + c + 3] = f32_to_bf16(o3);
}

// ---------------------------------------------------------------------------
// fp32 -> bf16 elementwise
// ---------------------------------------------------------------------------
__global__ __launch_bounds__(256) void f2bf_kernel(
    const float* __restrict__ in, unsigned short* __restrict__ out, int n)
{
    int i = blockIdx.x * 256 + threadIdx.x;
    if (i < n) out[i] = f32_to_bf16(in[i]);
}

// ---------------------------------------------------------------------------
// V [N,S,D] -> Vt [N,H,Dh,S]  (bf16)
// ---------------------------------------------------------------------------
__global__ __launch_bounds__(256) void vtrans_kernel(
    const unsigned short* __restrict__ V, unsigned short* __restrict__ Vt)
{
    size_t i = (size_t)blockIdx.x * 256 + threadIdx.x;   // over N*S*D
    int d = (int)(i & (D_HID - 1));
    size_t ns = i >> 10;               // n*S + s
    int s = (int)(ns & (SEQ - 1));
    int n = (int)(ns >> 11);
    int h = d >> 6, dh = d & 63;
    Vt[(((size_t)(n * NHEAD + h)) * DHEAD + dh) * SEQ + s] = V[i];
}

// ---------------------------------------------------------------------------
// GEMM: Y[M,Nn] = A[M,K](bf16) * B[Nn,K](bf16, torch-Linear weight) + bias.
// Block tile 128x128, 8 waves (2x4), each wave 64x32 (8 WMMA accumulators).
// Double-buffered async global->LDS staging (ASYNCcnt); 4 async-copy
// instructions per wave per tile, in-order completion -> wait asynccnt<=4.
// ---------------------------------------------------------------------------
__device__ __forceinline__ void stage_tile_async(
    const unsigned short* __restrict__ A, const unsigned short* __restrict__ B,
    unsigned short* As, unsigned short* Bs,
    int t, int bM, int bN, int K, int k0)
{
#pragma unroll
    for (int i = 0; i < 2; ++i) {
        int idx = t + i * 256;                     // 512 chunks of 16B each
        int row = idx >> 2, ch = (idx & 3) * 8;
        async_copy_b128(A + (size_t)(bM + row) * K + k0 + ch, As + row * 32 + ch);
        async_copy_b128(B + (size_t)(bN + row) * K + k0 + ch, Bs + row * 32 + ch);
    }
}

__global__ __launch_bounds__(256) void gemm_bf16_kernel(
    const unsigned short* __restrict__ A, const unsigned short* __restrict__ B,
    const float* __restrict__ bias, float* __restrict__ outF,
    unsigned short* __restrict__ outB, int M, int Nn, int K, int out_is_f32)
{
    __shared__ __align__(16) unsigned short As[2][128 * 32];
    __shared__ __align__(16) unsigned short Bs[2][128 * 32];

    const int t = threadIdx.x;
    const int wave = t >> 5, lane = t & 31;
    const int wm = wave >> 2, wn = wave & 3;           // 2 x 4 wave grid
    const int half = lane >> 4, l15 = lane & 15;
    const int bM = blockIdx.x * 128, bN = blockIdx.y * 128;

    v8f acc[4][2] = {};

    const int KT = K / 32;
    stage_tile_async(A, B, As[0], Bs[0], t, bM, bN, K, 0);

    for (int kt = 0; kt < KT; ++kt) {
        const int cur = kt & 1;
        if (kt + 1 < KT) {
            stage_tile_async(A, B, As[cur ^ 1], Bs[cur ^ 1], t, bM, bN, K,
                             (kt + 1) * 32);
            wait_asynccnt<4>();        // current tile's 4 copies retired
        } else {
            wait_asynccnt<0>();
        }
        __syncthreads();               // all waves' LDS writes visible

        Frag16 a[4], b[2];
#pragma unroll
        for (int mt = 0; mt < 4; ++mt) {
            int row = wm * 64 + mt * 16 + l15;
            const unsigned short* p = As[cur] + row * 32 + half * 8;
            a[mt].u[0] = *(const uint4*)p;
            a[mt].u[1] = *(const uint4*)(p + 16);
        }
#pragma unroll
        for (int nt = 0; nt < 2; ++nt) {
            int col = wn * 32 + nt * 16 + l15;
            const unsigned short* p = Bs[cur] + col * 32 + half * 8;
            b[nt].u[0] = *(const uint4*)p;
            b[nt].u[1] = *(const uint4*)(p + 16);
        }
#pragma unroll
        for (int mt = 0; mt < 4; ++mt)
#pragma unroll
            for (int nt = 0; nt < 2; ++nt)
                acc[mt][nt] = wmma_bf16(a[mt].v, b[nt].v, acc[mt][nt]);

        __syncthreads();               // done reading before buffer reuse
    }

#pragma unroll
    for (int mt = 0; mt < 4; ++mt)
#pragma unroll
        for (int nt = 0; nt < 2; ++nt) {
            int col = bN + wn * 32 + nt * 16 + l15;
            float bv = bias[col];
#pragma unroll
            for (int r = 0; r < 8; ++r) {
                int row = bM + wm * 64 + mt * 16 + half * 8 + r;
                float val = acc[mt][nt][r] + bv;
                if (out_is_f32) outF[(size_t)row * Nn + col] = val;
                else            outB[(size_t)row * Nn + col] = f32_to_bf16(val);
            }
        }
}

// ---------------------------------------------------------------------------
// Flash attention: 1 wave = one 16-row query tile of one (n,h).
// Q,K: [N,S,D] bf16 (head slice contiguous over Dh); Vt: [N,H,Dh,S] bf16.
// ---------------------------------------------------------------------------
__global__ __launch_bounds__(128) void flash_kernel(
    const unsigned short* __restrict__ Q, const unsigned short* __restrict__ Kb,
    const unsigned short* __restrict__ Vt, unsigned short* __restrict__ Z)
{
    __shared__ __align__(16) unsigned short P[4][16 * 32];

    const int wave = threadIdx.x >> 5, lane = threadIdx.x & 31;
    const int half = lane >> 4, l15 = lane & 15;

    const int tile = blockIdx.x * 4 + wave;           // 0 .. N*H*S/16-1
    const int n  = tile >> 11;                        // / (H * S/16) = /2048
    const int rm = tile & 2047;
    const int h  = rm >> 7;
    const int qs = (rm & 127) * 16;

    // Q A-fragments: dh 0..31 and 32..63
    Frag16 aQ[2];
    const unsigned short* qbase =
        Q + ((size_t)(n * SEQ + qs + l15)) * D_HID + h * DHEAD;
#pragma unroll
    for (int d = 0; d < 2; ++d) {
        int off = d * 32 + half * 8;
        aQ[d].u[0] = *(const uint4*)(qbase + off);
        aQ[d].u[1] = *(const uint4*)(qbase + off + 16);
    }

    float mstat[8], lstat[8];
    v8f O[4] = {};
#pragma unroll
    for (int r = 0; r < 8; ++r) { mstat[r] = -1e30f; lstat[r] = 0.0f; }

    const float scale = 0.125f;                        // 1/sqrt(64)
    const int kend = qs + 16;

    for (int kb = 0; kb < kend; kb += 32) {
        // S = Q * K^T  (two 16-col tiles, two K-steps over Dh each)
        v8f Sc[2] = {};
#pragma unroll
        for (int c = 0; c < 2; ++c) {
            const unsigned short* kbase =
                Kb + ((size_t)(n * SEQ + kb + c * 16 + l15)) * D_HID + h * DHEAD;
#pragma unroll
            for (int d = 0; d < 2; ++d) {
                Frag16 bK;
                const unsigned short* p = kbase + d * 32 + half * 8;
                bK.u[0] = *(const uint4*)p;
                bK.u[1] = *(const uint4*)(p + 16);
                Sc[c] = wmma_bf16(aQ[d].v, bK.v, Sc[c]);
            }
        }

        // online softmax per row (row = qs + half*8 + r; cols across 16 lanes)
#pragma unroll
        for (int r = 0; r < 8; ++r) {
            const int row = qs + half * 8 + r;
            float best = -1e30f;
#pragma unroll
            for (int c = 0; c < 2; ++c) {
                int col = kb + c * 16 + l15;
                float s = Sc[c][r] * scale;
                if (col > row) s = -1e30f;             // causal mask
                Sc[c][r] = s;
                best = fmaxf(best, s);
            }
#pragma unroll
            for (int m = 1; m < 16; m <<= 1)
                best = fmaxf(best, __shfl_xor(best, m, 16));
            float mnew = fmaxf(mstat[r], best);
            float alpha = __expf(mstat[r] - mnew);
            float psum = 0.0f;
#pragma unroll
            for (int c = 0; c < 2; ++c) {
                float p = __expf(Sc[c][r] - mnew);
                Sc[c][r] = p;
                psum += p;
            }
#pragma unroll
            for (int m = 1; m < 16; m <<= 1)
                psum += __shfl_xor(psum, m, 16);
            lstat[r] = lstat[r] * alpha + psum;
            mstat[r] = mnew;
#pragma unroll
            for (int j = 0; j < 4; ++j) O[j][r] *= alpha;
        }

        // stage P (bf16) through per-wave LDS to re-layout C -> A fragment
        unsigned short* pb = P[wave];
#pragma unroll
        for (int c = 0; c < 2; ++c)
#pragma unroll
            for (int r = 0; r < 8; ++r)
                pb[(half * 8 + r) * 32 + c * 16 + l15] = f32_to_bf16(Sc[c][r]);
        asm volatile("s_wait_dscnt 0x0" ::: "memory");  // same-wave DS RAW

        Frag16 aP;
        {
            const unsigned short* p = pb + l15 * 32 + half * 8;
            aP.u[0] = *(const uint4*)p;
            aP.u[1] = *(const uint4*)(p + 16);
        }

        // O += P * V  (4 dh tiles of 16)
#pragma unroll
        for (int j = 0; j < 4; ++j) {
            Frag16 bV;
            const unsigned short* p =
                Vt + (((size_t)(n * NHEAD + h)) * DHEAD + j * 16 + l15) * SEQ
                   + kb + half * 8;
            bV.u[0] = *(const uint4*)p;
            bV.u[1] = *(const uint4*)(p + 16);
            O[j] = wmma_bf16(aP.v, bV.v, O[j]);
        }
    }

    // finalize and store Z (bf16) [N,S,D]
#pragma unroll
    for (int r = 0; r < 8; ++r) {
        const float inv = 1.0f / lstat[r];
        const int row = qs + half * 8 + r;
#pragma unroll
        for (int j = 0; j < 4; ++j) {
            int col = h * DHEAD + j * 16 + l15;
            Z[(size_t)(n * SEQ + row) * D_HID + col] = f32_to_bf16(O[j][r] * inv);
        }
    }
}

// ---------------------------------------------------------------------------
extern "C" void kernel_launch(void* const* d_in, const int* in_sizes, int n_in,
                              void* d_out, int out_size, void* d_ws, size_t ws_size,
                              hipStream_t stream)
{
    const float* x   = (const float*)d_in[0];
    const float* Wq  = (const float*)d_in[1];
    const float* bq  = (const float*)d_in[2];
    const float* Wk  = (const float*)d_in[3];
    const float* bk  = (const float*)d_in[4];
    const float* Wv  = (const float*)d_in[5];
    const float* bv  = (const float*)d_in[6];
    const float* Wo  = (const float*)d_in[7];
    const float* bo  = (const float*)d_in[8];
    const float* lng = (const float*)d_in[9];
    const float* lnb = (const float*)d_in[10];
    float* out = (float*)d_out;

    const size_t NS  = (size_t)NBATCH * SEQ;          // 8192 rows
    const size_t NSD = NS * D_HID;                    // 8.4M elements
    const size_t W   = (size_t)D_HID * D_HID;         // 1M elements

    unsigned short* xn  = (unsigned short*)d_ws;
    unsigned short* wqb = xn  + NSD;
    unsigned short* wkb = wqb + W;
    unsigned short* wvb = wkb + W;
    unsigned short* wob = wvb + W;
    unsigned short* Qb  = wob + W;
    unsigned short* Kb  = Qb  + NSD;
    unsigned short* Vb  = Kb  + NSD;
    unsigned short* Vt  = Vb  + NSD;
    unsigned short* Zb  = Vt  + NSD;

    // 1) LayerNorm -> bf16
    ln_kernel<<<(int)NS, 256, 0, stream>>>(x, lng, lnb, xn);

    // 2) Weights -> bf16
    f2bf_kernel<<<(int)(W / 256), 256, 0, stream>>>(Wq, wqb, (int)W);
    f2bf_kernel<<<(int)(W / 256), 256, 0, stream>>>(Wk, wkb, (int)W);
    f2bf_kernel<<<(int)(W / 256), 256, 0, stream>>>(Wv, wvb, (int)W);
    f2bf_kernel<<<(int)(W / 256), 256, 0, stream>>>(Wo, wob, (int)W);

    // 3) Q/K/V projections (bf16 out)
    dim3 ggrid((int)(NS / 128), D_HID / 128);
    gemm_bf16_kernel<<<ggrid, 256, 0, stream>>>(xn, wqb, bq, nullptr, Qb,
                                                (int)NS, D_HID, D_HID, 0);
    gemm_bf16_kernel<<<ggrid, 256, 0, stream>>>(xn, wkb, bk, nullptr, Kb,
                                                (int)NS, D_HID, D_HID, 0);
    gemm_bf16_kernel<<<ggrid, 256, 0, stream>>>(xn, wvb, bv, nullptr, Vb,
                                                (int)NS, D_HID, D_HID, 0);

    // 4) V transpose -> [N,H,Dh,S]
    vtrans_kernel<<<(int)(NSD / 256), 256, 0, stream>>>(Vb, Vt);

    // 5) causal flash attention (1 wave per 16-row query tile, 4 waves/block)
    flash_kernel<<<(NBATCH * NHEAD * (SEQ / 16)) / 4, 128, 0, stream>>>(Qb, Kb, Vt, Zb);

    // 6) output projection (fp32 out)
    gemm_bf16_kernel<<<ggrid, 256, 0, stream>>>(Zb, wob, bo, out, nullptr,
                                                (int)NS, D_HID, D_HID, 1);

    (void)in_sizes; (void)n_in; (void)out_size; (void)ws_size;
}
